// RamseyGraphGNNWithCliqueAttention_90460601188723
// MI455X (gfx1250) — matrix operand order repaired and checked by reading, not compiled, not run on hardware
//
#include <hip/hip_runtime.h>
#include <hip/hip_bf16.h>

typedef __attribute__((ext_vector_type(16))) _Float16 v16h;
typedef __attribute__((ext_vector_type(8)))  _Float16 v8h;
typedef __attribute__((ext_vector_type(8)))  float    v8f;

#define DEVINL __device__ __forceinline__

// ---------------- parameter plumbing ----------------
// d_in layout (top-level insertion order; params pytree sorted):
// [0]=graph_vec, [1]=r_cliques, [2]=b_cliques,
// [3..34]=branch 'b' (sorted), [35]=emb_b, [36]=emb_w,
// [37..39]=gnn_b[0..2], [40..42]=gnn_w[0..2], [43..74]=branch 'r' (sorted)
enum {
  DOWN_B, DOWN_W, FF1_B, FF1_W, FF2_B, FF2_W, H1_B, H1_W, H2_B, H2_W,
  IN_K_B, IN_K_W, IN_Q_B, IN_Q_W, IN_V_B, IN_V_W, K_B, K_W,
  LN1_B, LN1_G, LN2_B, LN2_G, N2C_B, N2C_W, OUT_B, OUT_W,
  Q_B, Q_W, SIZE_B, SIZE_W, V_B, V_W, NBR
};
struct Branch  { const float* p[NBR]; };
struct BranchT {
  const _Float16 *qT, *kT, *vT, *iqT, *ikT, *ivT, *outT, *n2cT, *ff1T, *ff2T;
  const float* downT;   // f32 [64][1024]
};
struct Net {
  const float *emb_b, *emb_w;
  const float *gnn_b[3];
  const _Float16 *gT[3][2];   // per layer: Wtop^T, Wbot^T  (f16, [64][kpad])
  Branch  r,  b;
  BranchT rt, bt;
};

// ---------------- prep: transpose + convert weights into workspace ----------------
// dst[n*kpad + k] = (k < K) ? src[k*N + n] : 0
__global__ void prep_transpose(const float* __restrict__ src, void* __restrict__ dst,
                               int K, int N, int kpad, int asHalf) {
  const int idx = blockIdx.x * 256 + threadIdx.x;
  if (idx >= N * kpad) return;
  const int n = idx / kpad, k = idx % kpad;
  const float v = (k < K) ? src[k * N + n] : 0.0f;
  if (asHalf) ((_Float16*)dst)[idx] = (_Float16)v;
  else        ((float*)dst)[idx] = v;
}

// ---------------- WMMA fragment helpers (ISA 7.12.2 layouts, wave32) ----------------
DEVINL v16h fragA(const _Float16* src, int ld) {
  const int l = threadIdx.x & 31;
  const int m = l & 15, hi = l >> 4;
  const _Float16* p = src + m * ld + hi * 8;
  const v8h lo = *(const v8h*)p;
  const v8h hh = *(const v8h*)(p + 16);
  return __builtin_shufflevector(lo, hh, 0, 1, 2, 3, 4, 5, 6, 7,
                                 8, 9, 10, 11, 12, 13, 14, 15);
}

// B 32x16 tile where element (k,n) = baseT[n*stride + k]
DEVINL v16h fragB_T(const _Float16* baseT, int stride) {
  const int l = threadIdx.x & 31;
  const int n = l & 15, kb = (l >> 4) << 3;
  const _Float16* p = baseT + n * stride + kb;
  const v8h lo = *(const v8h*)p;
  const v8h hh = *(const v8h*)(p + 16);
  return __builtin_shufflevector(lo, hh, 0, 1, 2, 3, 4, 5, 6, 7,
                                 8, 9, 10, 11, 12, 13, 14, 15);
}

DEVINL v8f wmma_f16(v16h a, v16h b, v8f c) {
  return __builtin_amdgcn_wmma_f32_16x16x32_f16(false, a, false, b, (short)0, c, false, false);
}

// ---------------- unified tiled GEMM driver (fully compile-time specialized) ----------
// D = act( A0 @ B0 [+ A1 @ B1] + bias [+ res] )
// A*: row-major f16 (LDS). B*: transposed layout, stride sb (LDS f16 or global f16).
template<int MT, int NT, int KS, bool A1E, bool BIASE, bool RESE, bool RELU,
         bool DHE, bool DTE, bool DFE>
DEVINL void mmT(const _Float16* A0, int lda, const _Float16* B0, int sb,
                const _Float16* A1, const _Float16* B1, const float* bias,
                _Float16* dh, int ldd, _Float16* dt, int ldt,
                float* df, int ldf, const _Float16* res, int ldr) {
  const int l = threadIdx.x & 31;
  const int n = l & 15, hi = l >> 4;
#pragma unroll
  for (int mt = 0; mt < MT; ++mt) {
#pragma unroll
    for (int nt = 0; nt < NT; ++nt) {
      v8f c = {};
#pragma unroll
      for (int ks = 0; ks < KS; ++ks) {
        c = wmma_f16(fragA(A0 + mt * 16 * lda + ks * 32, lda),
                     fragB_T(B0 + nt * 16 * sb + ks * 32, sb), c);
        if (A1E)
          c = wmma_f16(fragA(A1 + mt * 16 * lda + ks * 32, lda),
                       fragB_T(B1 + nt * 16 * sb + ks * 32, sb), c);
      }
      const int gn = nt * 16 + n;
      const float bv = BIASE ? bias[gn] : 0.0f;
      float vals[8];
#pragma unroll
      for (int r = 0; r < 8; ++r) {
        const int m = mt * 16 + hi * 8 + r;
        float x = c[r] + bv;
        if (RESE) x += (float)res[m * ldr + gn];
        if (RELU) x = fmaxf(x, 0.0f);
        if (DHE) dh[m * ldd + gn] = (_Float16)x;
        if (DFE) df[m * ldf + gn] = x;
        vals[r] = x;
      }
      if (DTE) {
        v8h t;
#pragma unroll
        for (int r = 0; r < 8; ++r) t[r] = (_Float16)vals[r];
        *(v8h*)(dt + gn * ldt + mt * 16 + hi * 8) = t;
      }
    }
  }
}

// ---------------- model stages ----------------
DEVINL void cross_step(const Branch& p, const BranchT& pt, const int* __restrict__ cl,
                       const _Float16* xin, _Float16* xout, _Float16* xoutT,
                       _Float16* Ah, _Float16* T0h, _Float16* CT, int* CLs) {
  const int l = threadIdx.x & 31;
  for (int i = l; i < 64; i += 32) CLs[i] = cl[i];
  __syncthreads();
  for (int idx = l; idx < 1024; idx += 32) {
    const int c = idx >> 6, col = idx & 63;
    float s = 0.f;
#pragma unroll
    for (int t = 0; t < 4; ++t) s += (float)xin[CLs[c * 4 + t] * 64 + col];
    T0h[idx] = (_Float16)(0.25f * s);
  }
  __syncthreads();
  // crepsT = (xm @ n2c_w + b)^T -> CT [64][32] (cols 16..31 stay zero)
  mmT<1, 4, 2, false, true, false, false, false, true, false>(
      T0h, 64, pt.n2cT, 64, nullptr, nullptr, p.p[N2C_B],
      nullptr, 0, CT, 32, nullptr, 0, nullptr, 0);
  __syncthreads();
  if (l < 17) {
    float e[16];
    float sum = 0.f;
#pragma unroll
    for (int c = 0; c < 16; ++c) {
      int memb = 0;
#pragma unroll
      for (int t = 0; t < 4; ++t) memb |= (CLs[c * 4 + t] == l);
      const float ev = memb ? 2.718281828459045f : 1.0f;
      e[c] = ev; sum += ev;
    }
    const float inv = 1.0f / sum;
#pragma unroll
    for (int c = 0; c < 16; ++c) Ah[l * 32 + c] = (_Float16)(e[c] * inv);
#pragma unroll
    for (int c = 16; c < 32; ++c) Ah[l * 32 + c] = (_Float16)0.0f;
  }
  __syncthreads();
  // xout = xin + aw @ creps (dual store)
  mmT<2, 4, 1, false, false, true, false, true, true, false>(
      Ah, 32, CT, 32, nullptr, nullptr, nullptr,
      xout, 64, xoutT, 32, nullptr, 0, xin, 64);
  __syncthreads();
}

DEVINL void run_branch(const Branch& p, const BranchT& pt, const int* __restrict__ cl,
                       const _Float16* x2, float* Sf, _Float16* Ah, _Float16* W0h,
                       _Float16* T0h, _Float16* Qh, _Float16* Kh, _Float16* VT,
                       _Float16* Oh, _Float16* FFh, float* fscr, int* CLs,
                       float* __restrict__ outp) {
  const int l = threadIdx.x & 31;
  for (int i = l; i < 64; i += 32) CLs[i] = cl[i];
  __syncthreads();
  for (int idx = l; idx < 1024; idx += 32) {
    const int c = idx >> 6, col = idx & 63;
    float s = 0.f;
#pragma unroll
    for (int t = 0; t < 4; ++t) s += (float)x2[CLs[c * 4 + t] * 64 + col];
    W0h[idx] = (_Float16)(0.25f * s + 4.0f * p.p[SIZE_W][col] + p.p[SIZE_B][col]);
  }
  __syncthreads();
  // Q/K/V double projections
  mmT<1, 4, 2, false, true, false, false, true, false, false>(
      W0h, 64, pt.qT, 64, nullptr, nullptr, p.p[Q_B],
      T0h, 64, nullptr, 0, nullptr, 0, nullptr, 0);
  __syncthreads();
  mmT<1, 4, 2, false, true, false, false, true, false, false>(
      T0h, 64, pt.iqT, 64, nullptr, nullptr, p.p[IN_Q_B],
      Qh, 64, nullptr, 0, nullptr, 0, nullptr, 0);
  __syncthreads();
  mmT<1, 4, 2, false, true, false, false, true, false, false>(
      W0h, 64, pt.kT, 64, nullptr, nullptr, p.p[K_B],
      T0h, 64, nullptr, 0, nullptr, 0, nullptr, 0);
  __syncthreads();
  mmT<1, 4, 2, false, true, false, false, true, false, false>(
      T0h, 64, pt.ikT, 64, nullptr, nullptr, p.p[IN_K_B],
      Kh, 64, nullptr, 0, nullptr, 0, nullptr, 0);
  __syncthreads();
  mmT<1, 4, 2, false, true, false, false, true, false, false>(
      W0h, 64, pt.vT, 64, nullptr, nullptr, p.p[V_B],
      T0h, 64, nullptr, 0, nullptr, 0, nullptr, 0);
  __syncthreads();
  mmT<1, 4, 2, false, true, false, false, false, true, false>(
      T0h, 64, pt.ivT, 64, nullptr, nullptr, p.p[IN_V_B],
      nullptr, 0, VT, 32, nullptr, 0, nullptr, 0);
  __syncthreads();

  // 2 heads, DH=32: one 16x16x32 WMMA per score matrix
#pragma unroll
  for (int h0 = 0; h0 < 2; ++h0) {
    const int off = h0 * 32;
    {
      v16h a = fragA(Qh + off, 64);
      v16h b = fragB_T(Kh + off, 64);       // B[k][n] = K[n][off+k]
      v8f c = {};
      c = wmma_f16(a, b, c);
      const int n = l & 15, hi = l >> 4;
#pragma unroll
      for (int r = 0; r < 8; ++r)
        Sf[(hi * 8 + r) * 16 + n] = c[r] * 0.17677669529663687f;  // 1/sqrt(32)
    }
    __syncthreads();
    if (l < 16) {
      float mx = -1e30f;
#pragma unroll
      for (int k = 0; k < 16; ++k) mx = fmaxf(mx, Sf[l * 16 + k]);
      float e[16];
      float sum = 0.f;
#pragma unroll
      for (int k = 0; k < 16; ++k) { e[k] = __expf(Sf[l * 16 + k] - mx); sum += e[k]; }
      const float inv = 1.0f / sum;
#pragma unroll
      for (int k = 0; k < 16; ++k) Ah[l * 32 + k] = (_Float16)(e[k] * inv);
#pragma unroll
      for (int k = 16; k < 32; ++k) Ah[l * 32 + k] = (_Float16)0.0f;
    }
    __syncthreads();
    mmT<1, 2, 1, false, false, false, false, true, false, false>(
        Ah, 32, VT + off * 32, 32, nullptr, nullptr, nullptr,
        Oh + off, 64, nullptr, 0, nullptr, 0, nullptr, 0);
    __syncthreads();
  }

  // o @ out_w + out_b -> Sf f32 [16][64]
  mmT<1, 4, 2, false, true, false, false, false, false, true>(
      Oh, 64, pt.outT, 64, nullptr, nullptr, p.p[OUT_B],
      nullptr, 0, nullptr, 0, Sf, 64, nullptr, 0);
  __syncthreads();
  if (l < 16) {
    float m = 0.f;
    for (int c = 0; c < 64; ++c) m += Sf[l * 64 + c];
    m *= 0.015625f;
    float v = 0.f;
    for (int c = 0; c < 64; ++c) { const float d = Sf[l * 64 + c] - m; v += d * d; }
    v *= 0.015625f;
    const float rs = rsqrtf(v + 1e-5f);
    for (int c = 0; c < 64; ++c)
      Oh[l * 64 + c] = (_Float16)((Sf[l * 64 + c] - m) * rs * p.p[LN1_G][c]
                                  + p.p[LN1_B][c] + (float)W0h[l * 64 + c]);
  }
  __syncthreads();
  mmT<1, 16, 2, false, true, false, true, true, false, false>(
      Oh, 64, pt.ff1T, 64, nullptr, nullptr, p.p[FF1_B],
      FFh, 256, nullptr, 0, nullptr, 0, nullptr, 0);
  __syncthreads();
  mmT<1, 4, 8, false, true, false, false, false, false, true>(
      FFh, 256, pt.ff2T, 256, nullptr, nullptr, p.p[FF2_B],
      nullptr, 0, nullptr, 0, Sf, 64, nullptr, 0);
  __syncthreads();
  if (l < 16) {
    float m = 0.f;
    for (int c = 0; c < 64; ++c) m += Sf[l * 64 + c];
    m *= 0.015625f;
    float v = 0.f;
    for (int c = 0; c < 64; ++c) { const float d = Sf[l * 64 + c] - m; v += d * d; }
    v *= 0.015625f;
    const float rs = rsqrtf(v + 1e-5f);
    for (int c = 0; c < 64; ++c)
      Sf[l * 64 + c] = (Sf[l * 64 + c] - m) * rs * p.p[LN2_G][c]
                       + p.p[LN2_B][c] + (float)Oh[l * 64 + c];
  }
  __syncthreads();
  // d = a2.flatten(1024) @ down_w + down_b (transposed f32 copy, float4 loads)
  for (int n = l; n < 64; n += 32) {
    const float4* wp = (const float4*)(pt.downT + (size_t)n * 1024);
    float s = p.p[DOWN_B][n];
    for (int kk = 0; kk < 256; ++kk) {
      const float4 w4 = wp[kk];
      s += Sf[4 * kk] * w4.x + Sf[4 * kk + 1] * w4.y
         + Sf[4 * kk + 2] * w4.z + Sf[4 * kk + 3] * w4.w;
    }
    fscr[n] = s;
  }
  __syncthreads();
  for (int n = l; n < 64; n += 32) {
    float s = p.p[H1_B][n];
    for (int k = 0; k < 64; ++k) s += fscr[k] * p.p[H1_W][k * 64 + n];
    fscr[64 + n] = fmaxf(s, 0.f);
  }
  __syncthreads();
  float part = 0.f;
  for (int k = l; k < 64; k += 32) part += fscr[64 + k] * p.p[H2_W][k];
  Sf[l] = part;
  __syncthreads();
  if (l == 0) {
    float s = p.p[H2_B][0];
    for (int i = 0; i < 32; ++i) s += Sf[i];
    *outp = s;
  }
  __syncthreads();
}

// ---------------- main kernel: one wave32 per graph ----------------
__global__ __launch_bounds__(32)
void RamseyGraphGNNWithCliqueAttention_90460601188723_kernel(
    const float* __restrict__ gvecs, const int* __restrict__ rcl,
    const int* __restrict__ bcl, Net net, float* __restrict__ out) {
  __shared__ float    Sf[1024];
  __shared__ _Float16 Ah[1024];     // A_norm / aw / att (32x32)
  __shared__ _Float16 Xh[2048];     // x row-major (ping)
  __shared__ _Float16 XT[2048];     // x transposed [64][32] (ping)
  __shared__ _Float16 Zh[2048];     // pong
  __shared__ _Float16 ZT[2048];
  __shared__ _Float16 W0h[2048];    // A@x / cs
  __shared__ _Float16 T0h[1024];    // xm / tmp
  __shared__ _Float16 Qh[1024];
  __shared__ _Float16 Kh[1024];
  __shared__ _Float16 VT[2048];     // V transposed [64][32]
  __shared__ _Float16 CT[2048];     // creps transposed [64][32]
  __shared__ _Float16 Oh[1024];     // o / a1
  __shared__ _Float16 FFh[4096];    // 16x256
  __shared__ float    fscr[160];
  __shared__ int      CLs[64];

  const int g = blockIdx.x;
  const int l = threadIdx.x & 31;
  const float* __restrict__ gvec = gvecs + (size_t)g * 136;
  const int* __restrict__ rc = rcl + (size_t)g * 64;
  const int* __restrict__ bc = bcl + (size_t)g * 64;
  __builtin_prefetch(gvec, 0, 0);

  for (int i = l; i < 1024; i += 32) { Sf[i] = 0.f; Ah[i] = (_Float16)0.f; }
  for (int i = l; i < 2048; i += 32) {
    Xh[i] = (_Float16)0.f; XT[i] = (_Float16)0.f;
    Zh[i] = (_Float16)0.f; ZT[i] = (_Float16)0.f;
    W0h[i] = (_Float16)0.f; VT[i] = (_Float16)0.f; CT[i] = (_Float16)0.f;
  }
  __syncthreads();

  // ---- normalized adjacency -> Ah f16 [17][17] in 32x32 ----
  for (int e = l; e < 136; e += 32) {
    const float gv = gvec[e];
    int i = 0, rem = e;
    while (rem >= 16 - i) { rem -= 16 - i; ++i; }
    const int j = i + 1 + rem;
    Sf[i * 32 + j] = gv;
    Sf[j * 32 + i] = gv;
  }
  if (l < 17) Sf[l * 33] = 1.0f;
  __syncthreads();
  if (l < 17) {
    float s = 0.f;
    for (int j = 0; j < 17; ++j) s += Sf[l * 32 + j];
    fscr[l] = rsqrtf(s + 1e-8f);
  }
  __syncthreads();
  for (int idx = l; idx < 289; idx += 32) {
    const int i = idx / 17, j = idx % 17;
    Ah[i * 32 + j] = (_Float16)(fscr[i] * Sf[i * 32 + j] * fscr[j]);
  }
  for (int idx = l; idx < 272; idx += 32) {
    const int i = idx >> 4, c = idx & 15;
    const _Float16 v = (_Float16)(net.emb_w[c] + net.emb_b[c]);
    Xh[i * 64 + c] = v;
    XT[c * 32 + i] = v;
  }
  __syncthreads();

  // ---- GNN layer 0 (cin=16) ----
  _Float16 *xc = Xh, *xcT = XT, *xn = Zh, *xnT = ZT;
  mmT<2, 1, 1, false, false, false, false, true, false, false>(
      Ah, 32, xcT, 32, nullptr, nullptr, nullptr,
      W0h, 64, nullptr, 0, nullptr, 0, nullptr, 0);
  __syncthreads();
  mmT<2, 4, 1, true, true, false, true, true, true, false>(
      xc, 64, net.gT[0][0], 32, W0h, net.gT[0][1], net.gnn_b[0],
      xn, 64, xnT, 32, nullptr, 0, nullptr, 0);
  __syncthreads();
  { _Float16* t = xc; xc = xn; xn = t; t = xcT; xcT = xnT; xnT = t; }

  // ---- GNN layers 1,2 (cin=64) ----
  for (int L = 1; L < 3; ++L) {
    mmT<2, 4, 1, false, false, false, false, true, false, false>(
        Ah, 32, xcT, 32, nullptr, nullptr, nullptr,
        W0h, 64, nullptr, 0, nullptr, 0, nullptr, 0);
    __syncthreads();
    mmT<2, 4, 2, true, true, false, true, true, true, false>(
        xc, 64, net.gT[L][0], 64, W0h, net.gT[L][1], net.gnn_b[L],
        xn, 64, xnT, 32, nullptr, 0, nullptr, 0);
    __syncthreads();
    _Float16* t;
    t = xc; xc = xn; xn = t;
    t = xcT; xcT = xnT; xnT = t;
  }

  // ---- clique cross-attention (r then b) ----
  cross_step(net.r, net.rt, rc, xc, xn, xnT, Ah, T0h, CT, CLs);
  { _Float16* t = xc; xc = xn; xn = t; t = xcT; xcT = xnT; xnT = t; }
  cross_step(net.b, net.bt, bc, xc, xn, xnT, Ah, T0h, CT, CLs);
  { _Float16* t = xc; xc = xn; xn = t; t = xcT; xcT = xnT; xnT = t; }

  // ---- two scoring branches ----
  run_branch(net.r, net.rt, rc, xc, Sf, Ah, W0h, T0h, Qh, Kh, VT, Oh, FFh,
             fscr, CLs, out + (size_t)g * 2 + 0);
  run_branch(net.b, net.bt, bc, xc, Sf, Ah, W0h, T0h, Qh, Kh, VT, Oh, FFh,
             fscr, CLs, out + (size_t)g * 2 + 1);
}

// ---------------- host launch ----------------
static inline int cdiv(int a, int b) { return (a + b - 1) / b; }

extern "C" void kernel_launch(void* const* d_in, const int* in_sizes, int n_in,
                              void* d_out, int out_size, void* d_ws, size_t ws_size,
                              hipStream_t stream) {
  (void)out_size; (void)ws_size;
  auto F = [&](int i) -> const float* {
    if (i >= n_in) i = n_in - 1;
    return (const float*)d_in[i];
  };

  Net net;
  for (int i = 0; i < NBR; ++i) {
    net.b.p[i] = F(3 + i);
    net.r.p[i] = F(43 + i);
  }
  net.emb_b = F(35); net.emb_w = F(36);
  for (int i = 0; i < 3; ++i) net.gnn_b[i] = F(37 + i);
  const float* gw[3] = { F(40), F(41), F(42) };

  char* ws = (char*)d_ws;
  size_t off = 0;
  auto prep16 = [&](const float* src, int K, int N, int kpad) -> const _Float16* {
    _Float16* dst = (_Float16*)(ws + off);
    off = (off + (size_t)N * kpad * sizeof(_Float16) + 255) & ~(size_t)255;
    const int total = N * kpad;
    prep_transpose<<<cdiv(total, 256), 256, 0, stream>>>(src, dst, K, N, kpad, 1);
    return dst;
  };
  auto prep32 = [&](const float* src, int K, int N) -> const float* {
    float* dst = (float*)(ws + off);
    off = (off + (size_t)N * K * sizeof(float) + 255) & ~(size_t)255;
    const int total = N * K;
    prep_transpose<<<cdiv(total, 256), 256, 0, stream>>>(src, dst, K, N, K, 0);
    return dst;
  };

  int cins[3] = { 16, 64, 64 };
  for (int L = 0; L < 3; ++L) {
    const int kpad = (cins[L] < 32) ? 32 : cins[L];
    net.gT[L][0] = prep16(gw[L],                cins[L], 64, kpad);
    net.gT[L][1] = prep16(gw[L] + cins[L] * 64, cins[L], 64, kpad);
  }
  auto prepBranch = [&](const Branch& p) -> BranchT {
    BranchT t;
    t.qT   = prep16(p.p[Q_W],    64, 64, 64);
    t.kT   = prep16(p.p[K_W],    64, 64, 64);
    t.vT   = prep16(p.p[V_W],    64, 64, 64);
    t.iqT  = prep16(p.p[IN_Q_W], 64, 64, 64);
    t.ikT  = prep16(p.p[IN_K_W], 64, 64, 64);
    t.ivT  = prep16(p.p[IN_V_W], 64, 64, 64);
    t.outT = prep16(p.p[OUT_W],  64, 64, 64);
    t.n2cT = prep16(p.p[N2C_W],  64, 64, 64);
    t.ff1T = prep16(p.p[FF1_W],  64, 256, 64);
    t.ff2T = prep16(p.p[FF2_W],  256, 64, 256);
    t.downT = prep32(p.p[DOWN_W], 1024, 64);
    return t;
  };
  net.rt = prepBranch(net.r);
  net.bt = prepBranch(net.b);

  const int B = in_sizes[0] / 136;   // 32768
  if (B <= 0) return;
  RamseyGraphGNNWithCliqueAttention_90460601188723_kernel<<<B, 32, 0, stream>>>(
      (const float*)d_in[0], (const int*)d_in[1], (const int*)d_in[2],
      net, (float*)d_out);
}
